// BlockSparseAttention_54374285968047
// MI455X (gfx1250) — matrix operand order repaired and checked by reading, not compiled, not run on hardware
//
#include <hip/hip_runtime.h>
#include <stdint.h>

typedef __attribute__((ext_vector_type(16))) __bf16 v16bf;
typedef __attribute__((ext_vector_type(8)))  float  v8f;

struct B32x2 { uint4 a, b; };

__device__ __forceinline__ v16bf frag_cast(uint4 lo, uint4 hi) {
  B32x2 t{lo, hi};
  return __builtin_bit_cast(v16bf, t);
}

// bf16 RNE from f32
__device__ __forceinline__ unsigned short f2bf(float f) {
  unsigned int u = __builtin_bit_cast(unsigned int, f);
  unsigned int r = u + 0x7FFFu + ((u >> 16) & 1u);
  return (unsigned short)(r >> 16);
}

// A fragment (16x32 bf16, M x K): lane<16 -> row=lane, K in {k0..k0+7} u {k0+16..k0+23};
// lanes 16..31 -> same rows, K shifted +8. Matrix row-major, ld elements.
__device__ __forceinline__ v16bf load_fragA(const unsigned short* p, int row0, int k0, int ld) {
  int lane = threadIdx.x & 31;
  const unsigned short* q = p + (size_t)(row0 + (lane & 15)) * ld + k0 + ((lane >> 4) << 3);
  uint4 lo = *(const uint4*)q;
  uint4 hi = *(const uint4*)(q + 16);
  return frag_cast(lo, hi);
}

// B fragment (32x16 bf16, K x N): memory holds B^T row-major (N x K, ld elements).
// lane<16 -> col=lane, K in {k0..k0+7} u {k0+16..k0+23}; lanes 16..31 -> same cols, K shifted +8.
__device__ __forceinline__ v16bf load_fragB(const unsigned short* p, int n0, int k0, int ld) {
  int lane = threadIdx.x & 31;
  const unsigned short* q = p + (size_t)(n0 + (lane & 15)) * ld + k0 + ((lane >> 4) << 3);
  uint4 lo = *(const uint4*)q;
  uint4 hi = *(const uint4*)(q + 16);
  return frag_cast(lo, hi);
}

__device__ __forceinline__ v8f wmma_bf16(v16bf a, v16bf b, v8f c) {
  return __builtin_amdgcn_wmma_f32_16x16x32_bf16(false, a, false, b, (short)0, c, false, false);
}

// LDS byte offset (within the workgroup allocation) of a generic pointer to __shared__.
__device__ __forceinline__ unsigned int lds_off_u32(const void* p) {
  return (unsigned int)(unsigned long long)(__attribute__((address_space(3))) const void*)p;
}

// CDNA5 async copy global -> LDS, 16B per lane, tracked by ASYNCcnt.
__device__ __forceinline__ void async_b128(unsigned int lds_byte_off, const void* gaddr) {
  asm volatile("global_load_async_to_lds_b128 %0, %1, off"
               :: "v"(lds_byte_off), "v"(gaddr)
               : "memory");
}

__device__ __forceinline__ void wait_async0() {
  asm volatile("s_wait_asynccnt 0x0" ::: "memory");
}

// ---------------------------------------------------------------- fp32 -> bf16
__global__ void bsattn_cvt_bf16(const float* __restrict__ src,
                                unsigned short* __restrict__ dst, int n) {
  int i = (blockIdx.x * blockDim.x + threadIdx.x) * 4;
  if (i + 3 < n) {
    float4 f = *(const float4*)(src + i);
    dst[i + 0] = f2bf(f.x);
    dst[i + 1] = f2bf(f.y);
    dst[i + 2] = f2bf(f.z);
    dst[i + 3] = f2bf(f.w);
  }
}

// ---------------------------------------------------------------- QKV GEMM
// qkv[m,n] = sum_k x[m,k]*w_qkv[n,k] + b_qkv[n]; M=4096, N=3072, K=1024.
// Epilogue scatters into Q [H][L][64] (scaled by 0.125), K [H][L][64], V^T [H][64][L].
__global__ __launch_bounds__(256) void bsattn_qkv_gemm(
    const unsigned short* __restrict__ xb, const unsigned short* __restrict__ wb,
    const float* __restrict__ bias,
    unsigned short* __restrict__ Qh, unsigned short* __restrict__ Kh,
    unsigned short* __restrict__ VT) {
  const int L = 4096, D = 1024;
  int bm = blockIdx.y, bn = blockIdx.x;
  int w = threadIdx.x >> 5;
  int m0 = bm * 128 + (w & 3) * 32;
  int n0 = bn * 128 + (w >> 2) * 64;
  v8f acc[2][4] = {};
  for (int k = 0; k < D; k += 32) {
    v16bf a0 = load_fragA(xb, m0,      k, D);
    v16bf a1 = load_fragA(xb, m0 + 16, k, D);
#pragma unroll
    for (int t = 0; t < 4; ++t) {
      v16bf b = load_fragB(wb, n0 + t * 16, k, D);
      acc[0][t] = wmma_bf16(a0, b, acc[0][t]);
      acc[1][t] = wmma_bf16(a1, b, acc[1][t]);
    }
  }
  int lane = threadIdx.x & 31, ln = lane & 15, hi = lane >> 4;
#pragma unroll
  for (int mt = 0; mt < 2; ++mt)
#pragma unroll
    for (int t = 0; t < 4; ++t)
#pragma unroll
      for (int r = 0; r < 8; ++r) {
        int m = m0 + mt * 16 + r + hi * 8;
        int n = n0 + t * 16 + ln;
        float v = acc[mt][t][r] + bias[n];
        int sec = n >> 10;          // 0=Q, 1=K, 2=V
        int nn = n & 1023;
        int h = nn >> 6, c = nn & 63;
        if (sec == 0)      Qh[((size_t)(h * L + m) << 6) + c] = f2bf(v * 0.125f);
        else if (sec == 1) Kh[((size_t)(h * L + m) << 6) + c] = f2bf(v);
        else               VT[(size_t)(h * 64 + c) * L + m]   = f2bf(v);
      }
}

// ---------------------------------------------------------------- block-sparse flash attention
// One workgroup per (head, query-block). 8 waves, each owns 16 query rows.
// K / V^T blocks are staged once per workgroup into LDS with async copies.
__global__ __launch_bounds__(256) void bsattn_attn(
    const unsigned short* __restrict__ Qh, const unsigned short* __restrict__ Kh,
    const unsigned short* __restrict__ VT, unsigned short* __restrict__ aout) {
  const int L = 4096;
  // [0, 17408): P slabs (8 waves x 16 x 136), ALIASED with K tile (128 x 72);
  // [17408, 26112): V^T tile (64 x 136). All strides keep 16B alignment.
  __shared__ unsigned short plds[26112];
  int h = blockIdx.y, qb = blockIdx.x;
  int w = threadIdx.x >> 5;
  int lane = threadIdx.x & 31, ln = lane & 15, hi = lane >> 4;
  const unsigned short* qp = Qh + (size_t)h * L * 64;
  const unsigned short* kp = Kh + (size_t)h * L * 64;
  const unsigned short* vt = VT + (size_t)h * 64 * L;
  unsigned short* slab = plds + w * 16 * 136;
  const unsigned short* ktile = plds;
  const unsigned short* vtile = plds + 17408;
  unsigned int kbase = lds_off_u32(plds);
  unsigned int vbase = kbase + 17408u * 2u;
  int qrow = qb * 128 + w * 16;

  // Q fragments are invariant across key blocks: hoist into registers.
  v16bf qa0 = load_fragA(qp, qrow, 0, 64);
  v16bf qa1 = load_fragA(qp, qrow, 32, 64);

  float mrun[8], lrun[8];
  v8f oacc[4] = {};
#pragma unroll
  for (int r = 0; r < 8; ++r) { mrun[r] = -1e30f; lrun[r] = 0.f; }

  int kbs[32]; int nkb;
  if (qb == 0 || qb == 24) { nkb = 32; for (int i = 0; i < 32; ++i) kbs[i] = i; }
  else { kbs[0] = 0; kbs[1] = 24; kbs[2] = qb; nkb = 3; }

  for (int ib = 0; ib < nkb; ++ib) {
    int kb = kbs[ib];
    // Async-stage K block (128x64 -> stride 72) and V^T block (64x128 -> stride 136).
    // 2048 x 16B transfers, 8 per thread.
#pragma unroll
    for (int j = 0; j < 4; ++j) {
      int i = (int)threadIdx.x + 256 * j;
      int kr = i >> 3, ks8 = (i & 7) << 3;
      async_b128(kbase + (unsigned)(kr * 72 + ks8) * 2u,
                 kp + (size_t)(kb * 128 + kr) * 64 + ks8);
      int vr = i >> 4, vs8 = (i & 15) << 3;
      async_b128(vbase + (unsigned)(vr * 136 + vs8) * 2u,
                 vt + (size_t)vr * L + kb * 128 + vs8);
    }
    wait_async0();
    __syncthreads();

    // S = Q * K^T (B fragments from LDS K tile)
    v8f s[8] = {};
#pragma unroll
    for (int t = 0; t < 8; ++t) {
      v16bf b0 = load_fragB(ktile, t * 16, 0, 72);
      s[t] = wmma_bf16(qa0, b0, s[t]);
      v16bf b1 = load_fragB(ktile, t * 16, 32, 72);
      s[t] = wmma_bf16(qa1, b1, s[t]);
    }

    // Online softmax; P is kept in s[] registers (K tile still being read by
    // other waves, so no LDS writes yet). Lane tracks rows r (+8 upper half).
#pragma unroll
    for (int r = 0; r < 8; ++r) {
      float bmax = -1e30f;
#pragma unroll
      for (int t = 0; t < 8; ++t) bmax = fmaxf(bmax, s[t][r]);
#pragma unroll
      for (int msk = 1; msk < 16; msk <<= 1) bmax = fmaxf(bmax, __shfl_xor(bmax, msk, 32));
      float mnew = fmaxf(mrun[r], bmax);
      float alpha = __expf(mrun[r] - mnew);
      mrun[r] = mnew;
      float rs = 0.f;
#pragma unroll
      for (int t = 0; t < 8; ++t) {
        float p = __expf(s[t][r] - mnew);
        s[t][r] = p;
        rs += p;
      }
#pragma unroll
      for (int msk = 1; msk < 16; msk <<= 1) rs += __shfl_xor(rs, msk, 32);
      lrun[r] = lrun[r] * alpha + rs;
#pragma unroll
      for (int t = 0; t < 4; ++t) oacc[t][r] = oacc[t][r] * alpha;
    }
    __syncthreads();   // everyone done reading K tile; safe to overwrite with P

    // Spill P (bf16) to this wave's own LDS slab (same-wave LDS ops are in order).
#pragma unroll
    for (int t = 0; t < 8; ++t)
#pragma unroll
      for (int r = 0; r < 8; ++r)
        slab[(r + hi * 8) * 136 + t * 16 + ln] = f2bf(s[t][r]);

    // O += P * V  (A = P from own slab, B = V^T tile rows = V columns)
#pragma unroll
    for (int ks = 0; ks < 128; ks += 32) {
      v16bf a = load_fragA(slab, 0, ks, 136);
#pragma unroll
      for (int t = 0; t < 4; ++t) {
        v16bf b = load_fragB(vtile, t * 16, ks, 136);
        oacc[t] = wmma_bf16(a, b, oacc[t]);
      }
    }
    __syncthreads();   // before next block's async copies overwrite K/V/P regions
  }

#pragma unroll
  for (int t = 0; t < 4; ++t)
#pragma unroll
    for (int r = 0; r < 8; ++r) {
      int row = qrow + r + hi * 8;
      int col = h * 64 + t * 16 + ln;
      aout[(size_t)row * 1024 + col] = f2bf(oacc[t][r] / lrun[r]);
    }
}

// ---------------------------------------------------------------- output projection
// out[m,n] = sum_k attn[m,k]*w_out[n,k] + b_out[n]; M=4096, N=1024, K=1024; fp32 out.
__global__ __launch_bounds__(256) void bsattn_out_gemm(
    const unsigned short* __restrict__ ab, const unsigned short* __restrict__ wob,
    const float* __restrict__ bias, float* __restrict__ out) {
  const int D = 1024;
  int bm = blockIdx.y, bn = blockIdx.x;
  int w = threadIdx.x >> 5;
  int m0 = bm * 128 + (w & 3) * 32;
  int n0 = bn * 128 + (w >> 2) * 64;
  v8f acc[2][4] = {};
  for (int k = 0; k < D; k += 32) {
    v16bf a0 = load_fragA(ab, m0,      k, D);
    v16bf a1 = load_fragA(ab, m0 + 16, k, D);
#pragma unroll
    for (int t = 0; t < 4; ++t) {
      v16bf b = load_fragB(wob, n0 + t * 16, k, D);
      acc[0][t] = wmma_bf16(a0, b, acc[0][t]);
      acc[1][t] = wmma_bf16(a1, b, acc[1][t]);
    }
  }
  int lane = threadIdx.x & 31, ln = lane & 15, hi = lane >> 4;
#pragma unroll
  for (int mt = 0; mt < 2; ++mt)
#pragma unroll
    for (int t = 0; t < 4; ++t)
#pragma unroll
      for (int r = 0; r < 8; ++r) {
        int m = m0 + mt * 16 + r + hi * 8;
        int n = n0 + t * 16 + ln;
        out[(size_t)m * D + n] = acc[mt][t][r] + bias[n];
      }
}

// ----------------------------------------------------------------
extern "C" void kernel_launch(void* const* d_in, const int* in_sizes, int n_in,
                              void* d_out, int out_size, void* d_ws, size_t ws_size,
                              hipStream_t stream) {
  (void)in_sizes; (void)n_in; (void)out_size; (void)ws_size;
  const float* x     = (const float*)d_in[0];
  const float* w_qkv = (const float*)d_in[1];
  const float* b_qkv = (const float*)d_in[2];
  const float* w_out = (const float*)d_in[3];
  const float* b_out = (const float*)d_in[4];
  float* out = (float*)d_out;

  const int L = 4096, D = 1024;
  unsigned char* ws = (unsigned char*)d_ws;
  size_t off = 0;
  auto walloc = [&](size_t bytes) -> unsigned short* {
    unsigned short* p = (unsigned short*)(ws + off);
    off += (bytes + 255) & ~(size_t)255;
    return p;
  };
  unsigned short* xb   = walloc((size_t)L * D * 2);       // x bf16
  unsigned short* wqb  = walloc((size_t)3 * D * D * 2);   // w_qkv bf16
  unsigned short* wob  = walloc((size_t)D * D * 2);       // w_out bf16
  unsigned short* Qh   = walloc((size_t)L * D * 2);       // [H][L][64], pre-scaled
  unsigned short* Kh   = walloc((size_t)L * D * 2);       // [H][L][64]
  unsigned short* VT   = walloc((size_t)L * D * 2);       // [H][64][L]
  unsigned short* aout = walloc((size_t)L * D * 2);       // [L][D]

  int n1 = L * D;
  bsattn_cvt_bf16<<<(n1 / 4 + 255) / 256, 256, 0, stream>>>(x, xb, n1);
  int n2 = 3 * D * D;
  bsattn_cvt_bf16<<<(n2 / 4 + 255) / 256, 256, 0, stream>>>(w_qkv, wqb, n2);
  int n3 = D * D;
  bsattn_cvt_bf16<<<(n3 / 4 + 255) / 256, 256, 0, stream>>>(w_out, wob, n3);

  bsattn_qkv_gemm<<<dim3(24, 32), 256, 0, stream>>>(xb, wqb, b_qkv, Qh, Kh, VT);
  bsattn_attn<<<dim3(32, 16), 256, 0, stream>>>(Qh, Kh, VT, aout);
  bsattn_out_gemm<<<dim3(8, 32), 256, 0, stream>>>(aout, wob, b_out, out);
}
